// MultiheadAttention_20856361189489
// MI455X (gfx1250) — matrix operand order repaired and checked by reading, not compiled, not run on hardware
//
#include <hip/hip_runtime.h>
#include <hip/hip_bf16.h>

typedef __attribute__((ext_vector_type(16))) __bf16 v16bf;
typedef __attribute__((ext_vector_type(8)))  __bf16 v8bf;
typedef __attribute__((ext_vector_type(4)))  __bf16 v4bf;
typedef __attribute__((ext_vector_type(8)))  float  v8f;
typedef unsigned int u32x4 __attribute__((ext_vector_type(4)));
typedef int i32x4 __attribute__((ext_vector_type(4)));
typedef int i32x8 __attribute__((ext_vector_type(8)));

#define D_MODEL 1024
#define NHEAD   16
#define HDIM    64
#define SEQ     2048
#define BATCH   2
#define BS      (BATCH * SEQ)    // 4096 token rows
#define BHN     (BATCH * NHEAD)  // 32 (batch,head) pairs

#define WMMA_BF16(a, b, c) \
  __builtin_amdgcn_wmma_f32_16x16x32_bf16(false, (a), false, (b), (short)0, (c), false, false)

#if defined(__AMDGCN__) && __has_builtin(__builtin_amdgcn_tensor_load_to_lds)
#define HAVE_TDM 1
#else
#define HAVE_TDM 0
#endif

#if HAVE_TDM
// Issue one TDM 2-D tile load (bf16 elements) global -> LDS. Wave-level op:
// call from a single wave; completion tracked with TENSORcnt.
// D# fields per CDNA5 ISA ch.8: group0 = {count, lds_addr, global_addr, type},
// group1 = {data_size, tensor dims, tile dims, dim0 stride}. Groups 2/3 unused (2-D).
static __device__ __forceinline__ void tdm_load_2d_bf16(
    const __bf16* gsrc, const __bf16* ldsdst,
    unsigned tile_d0, unsigned tile_d1,
    unsigned tensor_d0, unsigned tensor_d1, unsigned stride_d0)
{
  const unsigned long long ga = (unsigned long long)(size_t)gsrc;
  const unsigned lds = (unsigned)(size_t)ldsdst;  // LDS aperture: low 32 bits = LDS byte addr
  u32x4 g0 = {0u, 0u, 0u, 0u};
  g0[0] = 1u;                                                  // count=1 (valid), user mode
  g0[1] = lds;                                                 // lds_addr (bytes)
  g0[2] = (unsigned)ga;                                        // global_addr[31:0]
  g0[3] = (unsigned)((ga >> 32) & 0x01ffffffu) | (2u << 30);   // global_addr[56:32] | type=2
  i32x8 g1 = {0, 0, 0, 0, 0, 0, 0, 0};
  g1[0] = (int)(1u << 16);                                     // data_size=1 -> 2 bytes
  g1[1] = (int)(tensor_d0 << 16);                              // tensor_dim0[15:0]
  g1[2] = (int)((tensor_d0 >> 16) | (tensor_d1 << 16));        // tensor_dim0[31:16] | tensor_dim1[15:0]
  g1[3] = (int)((tensor_d1 >> 16) | (tile_d0 << 16));          // tensor_dim1[31:16] | tile_dim0
  g1[4] = (int)tile_d1;                                        // tile_dim1 (tile_dim2=0)
  g1[5] = (int)stride_d0;                                      // tensor_dim0_stride[31:0]
  i32x4 z4 = {0, 0, 0, 0};
#if __clang_major__ >= 23
  i32x8 z8 = {0, 0, 0, 0, 0, 0, 0, 0};
  __builtin_amdgcn_tensor_load_to_lds(g0, g1, z4, z4, z8, 0);
#else
  __builtin_amdgcn_tensor_load_to_lds(g0, g1, z4, z4, 0);
#endif
}
#endif

// ---- WMMA fragment loaders (per CDNA5 ISA 7.12.2 layout tables) ----
// A operand 16x32 bf16: lane l: row = row0+(l&15), K = kb..kb+7 and kb+16..kb+23, kb=(l>>4)*8
static __device__ __forceinline__ v16bf frag_a(const __bf16* A, int lda, int row0, int k0, int lane) {
  const int r = lane & 15, kb = (lane >> 4) * 8;
  const __bf16* p = A + (size_t)(row0 + r) * lda + k0 + kb;
  v8bf lo = *(const v8bf*)(p);
  v8bf hi = *(const v8bf*)(p + 16);
  v16bf a;
#pragma unroll
  for (int i = 0; i < 8; ++i) { a[i] = lo[i]; a[i + 8] = hi[i]; }
  return a;
}

// B operand 32x16 bf16 column-major (Bmat row-major [N][K] => column n is row n):
// lane j: col = col0+(j&15), K = kb..kb+7 and kb+16..kb+23, kb=(j>>4)*8
static __device__ __forceinline__ v16bf frag_b(const __bf16* Bm, int ldb, int col0, int k0, int lane) {
  const int n = lane & 15, kb = (lane >> 4) * 8;
  const __bf16* p = Bm + (size_t)(col0 + n) * ldb + k0 + kb;
  v8bf lo = *(const v8bf*)(p);
  v8bf hi = *(const v8bf*)(p + 16);
  v16bf b;
#pragma unroll
  for (int i = 0; i < 8; ++i) { b[i] = lo[i]; b[i + 8] = hi[i]; }
  return b;
}

// ---- fp32 -> bf16 conversion ----
__global__ void cvt_f32_bf16(const float4* __restrict__ src, __bf16* __restrict__ dst, int n4) {
  int i = blockIdx.x * blockDim.x + threadIdx.x;
  if (i < n4) {
    float4 v = src[i];
    v4bf o = { (__bf16)v.x, (__bf16)v.y, (__bf16)v.z, (__bf16)v.w };
    *(v4bf*)(dst + (size_t)i * 4) = o;
  }
}

// ---- NT GEMM: C[M,N] = A[M,K] * Bw[N,K]^T + bias ----
// 4 waves/block, each wave a 64x64 tile (4x4 accumulators, 16 WMMA per 32-wide K step).
// MODE 0: scatter-store bf16 into Q[bh][s][d], K[bh][s][d], Vt[bh][d][s]
// MODE 1: fp32 store to Cout
template <int MODE>
__global__ __launch_bounds__(128) void gemm_nt_bf16(
    const __bf16* __restrict__ A, const __bf16* __restrict__ Bw,
    const float* __restrict__ bias, int N, int K,
    float* __restrict__ Cout, __bf16* __restrict__ Qb,
    __bf16* __restrict__ Kbf, __bf16* __restrict__ Vtb)
{
  const int lane = threadIdx.x & 31;
  const int wave = threadIdx.x >> 5;
  const int tilesN = N >> 7;
  const int bm = (blockIdx.x / tilesN) * 128 + (wave >> 1) * 64;
  const int bn = (blockIdx.x % tilesN) * 128 + (wave & 1) * 64;

  v8f acc[4][4] = {};
  for (int k0 = 0; k0 < K; k0 += 32) {
    if (k0 + 32 < K) {  // L2 prefetch hint for next K panel
      __builtin_prefetch(A + (size_t)bm * K + k0 + 32, 0, 1);
      __builtin_prefetch(Bw + (size_t)bn * K + k0 + 32, 0, 1);
    }
    v16bf a[4], b[4];
#pragma unroll
    for (int i = 0; i < 4; ++i) a[i] = frag_a(A, K, bm + 16 * i, k0, lane);
#pragma unroll
    for (int j = 0; j < 4; ++j) b[j] = frag_b(Bw, K, bn + 16 * j, k0, lane);
#pragma unroll
    for (int i = 0; i < 4; ++i)
#pragma unroll
      for (int j = 0; j < 4; ++j)
        acc[i][j] = WMMA_BF16(a[i], b[j], acc[i][j]);
  }

  // C layout: lane<16 vgpr r -> C[r, lane]; lane>=16 vgpr r -> C[8+r, lane-16]
  const int halfq = (lane >> 4) * 8, ln = lane & 15;
#pragma unroll
  for (int j = 0; j < 4; ++j) {
    const int col = bn + j * 16 + ln;
    const float bv = bias[col];
    // Q/K/V selector: uniform within each 16-aligned column tile
    const int h = col / 192, c = col % 192;
    const int tt = c >> 6, d = c & 63;
#pragma unroll
    for (int i = 0; i < 4; ++i) {
      const int row0 = bm + i * 16 + halfq;
      if (MODE == 1) {
        float* p = Cout + (size_t)row0 * N + col;
#pragma unroll
        for (int r = 0; r < 8; ++r) p[(size_t)r * N] = acc[i][j][r] + bv;
      } else {
        const int b_ = row0 >> 11, s0 = row0 & 2047;
        const int bh = b_ * NHEAD + h;
        if (tt == 0) {
          __bf16* p = Qb + ((size_t)bh * SEQ + s0) * HDIM + d;
#pragma unroll
          for (int r = 0; r < 8; ++r) p[r * HDIM] = (__bf16)(acc[i][j][r] + bv);
        } else if (tt == 1) {
          __bf16* p = Kbf + ((size_t)bh * SEQ + s0) * HDIM + d;
#pragma unroll
          for (int r = 0; r < 8; ++r) p[r * HDIM] = (__bf16)(acc[i][j][r] + bv);
        } else {  // V stored transposed: Vt[bh][d][s]
          __bf16* p = Vtb + ((size_t)bh * HDIM + d) * SEQ + s0;
#pragma unroll
          for (int r = 0; r < 8; ++r) p[r] = (__bf16)(acc[i][j][r] + bv);
        }
      }
    }
  }
}

// ---- Flash attention, S^T formulation ----
// Block = 8 waves; block owns (bh, 128 queries); wave owns 16 queries.
// S^T = K*Q^T so each lane pair (j, j+16) owns query j&15: softmax state is one
// scalar per lane, row reductions are in-lane + ONE shfl_xor(16).
// O^T = V^T * P^T accumulated in registers.
// K/V tiles double-buffered in LDS; staged by the Tensor Data Mover when the
// toolchain exposes it (TENSORcnt pipeline), else a register-pipelined fallback.
__global__ __launch_bounds__(256) void attn_fwd(
    const __bf16* __restrict__ Qb, const __bf16* __restrict__ Kb,
    const __bf16* __restrict__ Vtb, __bf16* __restrict__ Ob)
{
  __shared__ __bf16 Ks[2][32 * 64];   // [key][d], double-buffered
  __shared__ __bf16 Vts[2][64 * 32];  // [d][key], double-buffered
  __shared__ __bf16 Pq[8][16 * 32];   // per-wave P, [query][key]

  const int lane = threadIdx.x & 31;
  const int wave = threadIdx.x >> 5;
  const int bh  = blockIdx.x >> 4;
  const int q0b = (blockIdx.x & 15) * 128;
  const int q0w = q0b + wave * 16;

  const __bf16* Qh  = Qb  + (size_t)bh * SEQ * HDIM;
  const __bf16* Kh  = Kb  + (size_t)bh * SEQ * HDIM;
  const __bf16* Vth = Vtb + (size_t)bh * HDIM * SEQ;

  // Q as B-operand (column n = query q0w+n, K = head dim)
  const v16bf bq0 = frag_b(Qh, HDIM, q0w, 0, lane);
  const v16bf bq1 = frag_b(Qh, HDIM, q0w, 32, lane);

  const int qmine = q0w + (lane & 15);  // query owned by this lane pair
  const int half8 = (lane >> 4) * 8;

  float m = -3.0e38f, l = 0.f;
  v8f accO[4] = {};

  const int kbEnd = q0b + 128;

#if HAVE_TDM
  if (wave == 0) {  // stage first tile pair via TDM (descriptor values all uniform)
    tdm_load_2d_bf16(Kh,  &Ks[0][0],  HDIM, 32, HDIM, SEQ, HDIM);  // 32 keys x 64 d
    tdm_load_2d_bf16(Vth, &Vts[0][0], 32, HDIM, SEQ, HDIM, SEQ);   // 64 d x 32 keys
  }
#else
  // fallback: cooperative staging, 256 threads x one 16B chunk per tile
  const int tid = threadIdx.x;
  const int krr = tid >> 3, ko = (tid & 7) * 8;  // K tile: 32 rows x 8 chunks
  const int drr = tid >> 2, dc = (tid & 3) * 8;  // Vt tile: 64 rows x 4 chunks
  v8bf rK = *(const v8bf*)&Kh[(size_t)krr * HDIM + ko];
  v8bf rV = *(const v8bf*)&Vth[(size_t)drr * SEQ + dc];
#endif

  int p = 0;
  for (int kb = 0; kb < kbEnd; kb += 32) {
#if HAVE_TDM
    if (wave == 0) __builtin_amdgcn_s_wait_tensorcnt(0);  // tile p landed
    __syncthreads();  // tile visible to all; previous readers of buf p^1 done
    if ((wave == 0) && (kb + 32 < kbEnd)) {  // DMA next tile into other buffer
      tdm_load_2d_bf16(Kh + (size_t)(kb + 32) * HDIM, &Ks[p ^ 1][0],
                       HDIM, 32, HDIM, SEQ, HDIM);
      tdm_load_2d_bf16(Vth + (kb + 32), &Vts[p ^ 1][0],
                       32, HDIM, SEQ, HDIM, SEQ);
    }
#else
    *(v8bf*)&Ks[p][krr * 64 + ko]  = rK;
    *(v8bf*)&Vts[p][drr * 32 + dc] = rV;
    __syncthreads();  // tile visible (other buffer free: single barrier)
    if (kb + 32 < kbEnd) {  // issue next tile's loads; waited at next ds_store
      rK = *(const v8bf*)&Kh[(size_t)(kb + 32 + krr) * HDIM + ko];
      rV = *(const v8bf*)&Vth[(size_t)drr * SEQ + (kb + 32) + dc];
    }
#endif

    if (kb < q0w + 16) {  // causal: wave-uniform -> EXEC stays all-ones for WMMA
      // S^T = K_blk * Q^T : lane holds 16 scores (keys) of its own query
      v8f st[2];
#pragma unroll
      for (int sub = 0; sub < 2; ++sub) {
        v8f s = {};
        v16bf ak0 = frag_a(&Ks[p][0], 64, sub * 16, 0, lane);
        v16bf ak1 = frag_a(&Ks[p][0], 64, sub * 16, 32, lane);
        s = WMMA_BF16(ak0, bq0, s);
        s = WMMA_BF16(ak1, bq1, s);
        st[sub] = s;
      }
      // scale + causal mask + in-lane block max
      float bmax = -3.0e38f;
#pragma unroll
      for (int sub = 0; sub < 2; ++sub)
#pragma unroll
        for (int r = 0; r < 8; ++r) {
          const int key = kb + sub * 16 + half8 + r;
          float v = st[sub][r] * 0.125f;  // 1/sqrt(64)
          if (key > qmine) v = -3.0e38f;
          st[sub][r] = v;
          bmax = fmaxf(bmax, v);
        }
      bmax = fmaxf(bmax, __shfl_xor(bmax, 16, 32));  // merge key halves
      const float mn = fmaxf(m, bmax);
      const float corr = __expf(m - mn);
      m = mn;
      float rs = 0.f;
#pragma unroll
      for (int sub = 0; sub < 2; ++sub) {
        v8bf pb;
#pragma unroll
        for (int r = 0; r < 8; ++r) {
          const float pe = __expf(st[sub][r] - mn);
          rs += pe;
          pb[r] = (__bf16)pe;
        }
        // P[query][key]: this lane's 8 keys are contiguous -> one 16B store
        *(v8bf*)&Pq[wave][(lane & 15) * 32 + sub * 16 + half8] = pb;
      }
      rs += __shfl_xor(rs, 16, 32);
      l = l * corr + rs;
#pragma unroll
      for (int t4 = 0; t4 < 4; ++t4)
#pragma unroll
        for (int r = 0; r < 8; ++r) accO[t4][r] *= corr;

      // O^T += V^T * P^T  (A = Vt rows from LDS, B = P rows = per-query keys)
      const v16bf bp = frag_b(&Pq[wave][0], 32, 0, 0, lane);
#pragma unroll
      for (int t4 = 0; t4 < 4; ++t4) {
        v16bf av = frag_a(&Vts[p][0], 32, t4 * 16, 0, lane);
        accO[t4] = WMMA_BF16(av, bp, accO[t4]);
      }
    }
    p ^= 1;
  }

  // O^T layout: lane j, vgpr r -> O[d = t4*16 + half8 + r][query = j&15].
  // Per lane: fixed query row, 8 contiguous d per tile -> 16B stores.
  const float inv = 1.0f / l;
  __bf16* orow = Ob + ((size_t)bh * SEQ + qmine) * HDIM;
#pragma unroll
  for (int t4 = 0; t4 < 4; ++t4) {
    v8bf ob;
#pragma unroll
    for (int r = 0; r < 8; ++r) ob[r] = (__bf16)(accO[t4][r] * inv);
    *(v8bf*)&orow[t4 * 16 + half8] = ob;
  }
}

extern "C" void kernel_launch(void* const* d_in, const int* in_sizes, int n_in,
                              void* d_out, int out_size, void* d_ws, size_t ws_size,
                              hipStream_t stream) {
  const float* x    = (const float*)d_in[0];
  const float* Wqkv = (const float*)d_in[1];
  const float* bqkv = (const float*)d_in[2];
  const float* Wo   = (const float*)d_in[3];
  const float* bo   = (const float*)d_in[4];
  float* out = (float*)d_out;

  char* ws = (char*)d_ws;
  const size_t MB = (size_t)1 << 20;
  __bf16* xb    = (__bf16*)(ws + 0 * MB);   // 8 MB  [4096,1024]
  __bf16* Wqkvb = (__bf16*)(ws + 8 * MB);   // 6 MB  [3072,1024]
  __bf16* Wob   = (__bf16*)(ws + 14 * MB);  // 2 MB  [1024,1024]
  __bf16* Qb    = (__bf16*)(ws + 16 * MB);  // 8 MB  [32,2048,64]
  __bf16* Kb    = (__bf16*)(ws + 24 * MB);  // 8 MB  [32,2048,64]
  __bf16* Vtb   = (__bf16*)(ws + 32 * MB);  // 8 MB  [32,64,2048] (V transposed)
  __bf16* Ob    = (__bf16*)(ws + 40 * MB);  // 8 MB  [32,2048,64] == flat [4096,1024]

  {
    int n4 = BS * D_MODEL / 4;
    cvt_f32_bf16<<<(n4 + 255) / 256, 256, 0, stream>>>((const float4*)x, xb, n4);
    n4 = 3 * D_MODEL * D_MODEL / 4;
    cvt_f32_bf16<<<(n4 + 255) / 256, 256, 0, stream>>>((const float4*)Wqkv, Wqkvb, n4);
    n4 = D_MODEL * D_MODEL / 4;
    cvt_f32_bf16<<<(n4 + 255) / 256, 256, 0, stream>>>((const float4*)Wo, Wob, n4);
  }

  // QKV projection: [4096,1024] x [3072,1024]^T, scatter into Q/K/V^T
  gemm_nt_bf16<0><<<(BS / 128) * (3 * D_MODEL / 128), 128, 0, stream>>>(
      xb, Wqkvb, bqkv, 3 * D_MODEL, D_MODEL, nullptr, Qb, Kb, Vtb);

  // Flash attention: 32 (b,h) pairs x 16 query blocks of 128
  attn_fwd<<<BHN * (SEQ / 128), 256, 0, stream>>>(Qb, Kb, Vtb, Ob);

  // Output projection: [4096,1024] x [1024,1024]^T -> fp32 out
  gemm_nt_bf16<1><<<(BS / 128) * (D_MODEL / 128), 128, 0, stream>>>(
      Ob, Wob, bo, D_MODEL, D_MODEL, out, nullptr, nullptr, nullptr);
}
